// NystromAttention_85375359910229
// MI455X (gfx1250) — matrix operand order repaired and checked by reading, not compiled, not run on hardware
//
#include <hip/hip_runtime.h>

typedef __attribute__((ext_vector_type(16))) _Float16 v16h;
typedef __attribute__((ext_vector_type(8)))  _Float16 v8h;
typedef __attribute__((ext_vector_type(8)))  float    v8f;
typedef __attribute__((ext_vector_type(2)))  float    v2f;

#define DEV __device__ __forceinline__

// ---------------------------------------------------------------------------
// WMMA helpers (gfx1250 wave32 layouts, CDNA5 ISA 7.12.2)
// ---------------------------------------------------------------------------
DEV v8f wmma16(v16h a, v16h b, v8f c) {
  return __builtin_amdgcn_wmma_f32_16x16x32_f16(false, a, false, b, (short)0, c,
                                                false, false);
}

// f32 WMMA: 16x16x4, A/B = 2 VGPRs per lane (ISA: lanes 0-15 K={0,1}, 16-31 K={2,3})
DEV v8f wmma4(v2f a, v2f b, v8f c) {
  return __builtin_amdgcn_wmma_f32_16x16x4_f32(false, a, false, b, (short)0, c,
                                               false, false);
}

// A fragment: 16x32 f16 tile, row-major source, row = lane&15.
// lanes 0-15 hold K = 0..15, lanes 16-31 hold K = 16..31.
DEV v16h load_a_frag(const _Float16* __restrict__ A0, int lda, int k0, int lane) {
  int r = lane & 15, hi = lane >> 4;
  const _Float16* p = A0 + (size_t)r * lda + k0 + hi * 16;
  v8h a0 = *(const v8h*)p;
  v8h a1 = *(const v8h*)(p + 8);
  v16h f;
#pragma unroll
  for (int i = 0; i < 8; ++i) { f[i] = a0[i]; f[i + 8] = a1[i]; }
  return f;
}

// B fragment for NT GEMM: operand stored row-major [N, K]; logical B[k][n] = Bst[n][k].
// lane col = lane&15, lanes 0-15 K=0..15, lanes 16-31 K=16..31 -> one contiguous 32B load.
DEV v16h load_bt_frag(const _Float16* __restrict__ B0, int ldb, int k0, int lane) {
  int c = lane & 15, hi = lane >> 4;
  return *(const v16h*)(B0 + (size_t)c * ldb + k0 + hi * 16);
}

// ---------------------------------------------------------------------------
// Elementwise / conversion kernels
// ---------------------------------------------------------------------------
__global__ void f32_to_f16_kernel(const float* __restrict__ in,
                                  _Float16* __restrict__ out, int n) {
  int i = blockIdx.x * 256 + threadIdx.x;
  if (i < n) out[i] = (_Float16)in[i];
}

__global__ void transpose_f32_to_f16_kernel(const float* __restrict__ in,
                                            _Float16* __restrict__ out,
                                            int rows, int cols) {
  int i = blockIdx.x * 256 + threadIdx.x;
  if (i < rows * cols) {
    int r = i / cols, c = i % cols;
    out[(size_t)c * rows + r] = (_Float16)in[i];
  }
}

__global__ void init_scal_kernel(float* __restrict__ s) {
  if (threadIdx.x < 2) s[threadIdx.x] = 0.0f;
}

// landmark mean over 16 consecutive rows: [bh,4096,64] -> [bh,256,64]
__global__ void landmark_kernel(const _Float16* __restrict__ src,
                                _Float16* __restrict__ dst) {
  int id = blockIdx.x * 256 + threadIdx.x;  // 32*256*64 total
  int d = id & 63, m = (id >> 6) & 255, bh = id >> 14;
  const _Float16* p = src + ((size_t)bh * 4096 + m * 16) * 64 + d;
  float s = 0.f;
#pragma unroll
  for (int i = 0; i < 16; ++i) s += (float)p[i * 64];
  dst[((size_t)bh * 256 + m) * 64 + d] = (_Float16)(s * (1.0f / 16.0f));
}

// out = coef*in + diag*I  (256x256 f32, batched over z)
__global__ void diag_axpy_f32_kernel(const float* __restrict__ in,
                                     float* __restrict__ out, float coef,
                                     float diag) {
  size_t base = (size_t)blockIdx.z * 65536;
  for (int idx = threadIdx.x; idx < 65536; idx += 256) {
    int i = idx >> 8, j = idx & 255;
    out[base + idx] = coef * in[base + idx] + ((i == j) ? diag : 0.f);
  }
}

// row/col-sum maxima of attn2 (all entries >= 0): scal[0]=max rowsum, scal[1]=max colsum
__global__ void attn2_reduce_kernel(const float* __restrict__ a2f,
                                    float* __restrict__ scal) {
  __shared__ float sred[256];
  size_t base = (size_t)blockIdx.x * 65536;
  int t = threadIdx.x;
  float rowsum = 0.f, colsum = 0.f;
  for (int k = 0; k < 256; ++k) {
    rowsum += a2f[base + (size_t)t * 256 + k];
    colsum += a2f[base + (size_t)k * 256 + t];
  }
  sred[t] = rowsum; __syncthreads();
  for (int off = 128; off; off >>= 1) {
    if (t < off) sred[t] = fmaxf(sred[t], sred[t + off]);
    __syncthreads();
  }
  if (!t) atomicMax((unsigned int*)&scal[0], __float_as_uint(sred[0]));
  __syncthreads();
  sred[t] = colsum; __syncthreads();
  for (int off = 128; off; off >>= 1) {
    if (t < off) sred[t] = fmaxf(sred[t], sred[t + off]);
    __syncthreads();
  }
  if (!t) atomicMax((unsigned int*)&scal[1], __float_as_uint(sred[0]));
}

// z0 = attn2^T / (maxrow*maxcol); writes z (row-major f32) and zT (f32)
__global__ void pinv_init_f32_kernel(const float* __restrict__ a2f,
                                     const float* __restrict__ scal,
                                     float* __restrict__ z,
                                     float* __restrict__ zT) {
  size_t base = (size_t)blockIdx.z * 65536;
  float inv = 1.0f / (scal[0] * scal[1]);
  for (int idx = threadIdx.x; idx < 65536; idx += 256) {
    int i = idx >> 8, j = idx & 255;
    float x = a2f[base + idx] * inv;     // attn2[i][j]/denom
    zT[base + idx] = x;                  // zT[i][j] = z[j][i]
    z[base + (size_t)j * 256 + i] = x;   // z[j][i]
  }
}

// depthwise conv residual + reshape to [b, n, h*64+d] f16
__global__ void residual_kernel(const float* __restrict__ O,
                                const _Float16* __restrict__ vh,
                                const float* __restrict__ Wres,
                                _Float16* __restrict__ hidden) {
  size_t id = (size_t)blockIdx.x * 256 + threadIdx.x;  // 32*4096*64
  int d = (int)(id & 63);
  int n = (int)((id >> 6) & 4095);
  int bh = (int)(id >> 18);
  int h = bh & 7, b = bh >> 3;
  float acc = O[id];
#pragma unroll
  for (int t = 0; t < 33; ++t) {
    int nn = n + t - 16;
    if (nn >= 0 && nn < 4096)
      acc += Wres[h * 33 + t] * (float)vh[((size_t)bh * 4096 + nn) * 64 + d];
  }
  hidden[((size_t)(b * 4096 + n)) * 512 + h * 64 + d] = (_Float16)acc;
}

// ---------------------------------------------------------------------------
// QKV GEMM: C = x_f16 @ WqkvT^T, scatter-store into q(scaled)/k/v/vT  (NT form)
// ---------------------------------------------------------------------------
__global__ __launch_bounds__(256) void gemm_qkv_kernel(
    const _Float16* __restrict__ X, const _Float16* __restrict__ WT,
    _Float16* __restrict__ qh, _Float16* __restrict__ kh,
    _Float16* __restrict__ vh, _Float16* __restrict__ vTh) {
  int tid = threadIdx.x, w = tid >> 5, lane = tid & 31;
  int c = lane & 15, hi = lane >> 4;
  const _Float16* Ab = X + (size_t)(blockIdx.y * 128 + w * 16) * 512;
  const _Float16* Bb = WT + (size_t)(blockIdx.x * 64) * 512;
  v8f acc[4] = {};
  for (int k0 = 0; k0 < 512; k0 += 32) {
    if (k0 + 32 < 512)
      __builtin_prefetch(Ab + (size_t)(lane & 15) * 512 + k0 + 32, 0, 3);
    v16h af = load_a_frag(Ab, 512, k0, lane);
#pragma unroll
    for (int t = 0; t < 4; ++t)
      acc[t] = wmma16(af, load_bt_frag(Bb + (size_t)(t * 16) * 512, 512, k0, lane),
                      acc[t]);
  }
#pragma unroll
  for (int t = 0; t < 4; ++t) {
    int colg = blockIdx.x * 64 + t * 16 + c;
    int which = colg >> 9, hd = colg & 511;
    int h = hd >> 6, d = hd & 63;
#pragma unroll
    for (int i = 0; i < 8; ++i) {
      int m = blockIdx.y * 128 + w * 16 + i + 8 * hi;
      int b = m >> 12, n = m & 4095;
      float v = acc[t][i];
      size_t o = ((size_t)(b * 8 + h) * 4096 + n) * 64 + d;
      if (which == 0)      qh[o] = (_Float16)(v * 0.125f);  // SCALE = 64^-0.5
      else if (which == 1) kh[o] = (_Float16)v;
      else {
        vh[o] = (_Float16)v;
        vTh[((size_t)(b * 8 + h) * 64 + d) * 4096 + n] = (_Float16)v;
      }
    }
  }
}

// ---------------------------------------------------------------------------
// Generic batched NT GEMM: C = A @ B^T (+bias).  Block tile 128x64 (8 waves x 4 tiles).
// ---------------------------------------------------------------------------
template <typename OutT, bool BIAS>
__global__ __launch_bounds__(256) void gemm_nt_kernel(
    const _Float16* __restrict__ A, int lda, long long sA,
    const _Float16* __restrict__ B, int ldb, long long sB,
    OutT* __restrict__ C, int ldc, long long sC, int K,
    const float* __restrict__ bias) {
  int tid = threadIdx.x, w = tid >> 5, lane = tid & 31;
  int c = lane & 15, hi = lane >> 4;
  size_t bh = blockIdx.z;
  const _Float16* Ab = A + bh * sA + (size_t)(blockIdx.y * 128 + w * 16) * lda;
  const _Float16* Bb = B + bh * sB + (size_t)(blockIdx.x * 64) * ldb;
  v8f acc[4] = {};
  for (int k0 = 0; k0 < K; k0 += 32) {
    if (k0 + 32 < K)
      __builtin_prefetch(Ab + (size_t)(lane & 15) * lda + k0 + 32, 0, 3);
    v16h af = load_a_frag(Ab, lda, k0, lane);
#pragma unroll
    for (int t = 0; t < 4; ++t)
      acc[t] = wmma16(af, load_bt_frag(Bb + (size_t)(t * 16) * ldb, ldb, k0, lane),
                      acc[t]);
  }
  OutT* Cb = C + bh * sC;
#pragma unroll
  for (int t = 0; t < 4; ++t) {
    int cg = blockIdx.x * 64 + t * 16 + c;
    float bv = BIAS ? bias[cg] : 0.f;
#pragma unroll
    for (int i = 0; i < 8; ++i) {
      int rg = blockIdx.y * 128 + w * 16 + i + 8 * hi;
      Cb[(size_t)rg * ldc + cg] = (OutT)(acc[t][i] + bv);
    }
  }
}

// ---------------------------------------------------------------------------
// 256x256 f32 NT matmul for pinv (V_WMMA_F32_16X16X4_F32):
// C = coef*(A@Bt^T) + diag*I, dual (normal / transposed) store.
// Block tile 32x64 (8 waves, 2x4).
// ---------------------------------------------------------------------------
__global__ __launch_bounds__(256) void mm256_f32_kernel(
    const float* __restrict__ A, const float* __restrict__ Bt,
    float* __restrict__ outN, float* __restrict__ outT,
    float coef, float diag) {
  int tid = threadIdx.x, w = tid >> 5, lane = tid & 31;
  int c = lane & 15, hi = lane >> 4;
  size_t base = (size_t)blockIdx.z * 65536;
  int wm = w >> 2, wn = w & 3;
  int row0 = blockIdx.y * 32 + wm * 16, col0 = blockIdx.x * 64 + wn * 16;
  const float* Ab = A + base + (size_t)(row0 + c) * 256;   // row = lane&15
  const float* Bb = Bt + base + (size_t)(col0 + c) * 256;  // col = lane&15
  v8f acc = {};
#pragma unroll 8
  for (int k0 = 0; k0 < 256; k0 += 4) {
    v2f a = *(const v2f*)(Ab + k0 + 2 * hi);
    v2f b = *(const v2f*)(Bb + k0 + 2 * hi);
    acc = wmma4(a, b, acc);
  }
#pragma unroll
  for (int i = 0; i < 8; ++i) {
    int rg = row0 + i + 8 * hi, cg = col0 + c;
    float v = coef * acc[i] + ((rg == cg) ? diag : 0.f);
    if (outN) outN[base + (size_t)rg * 256 + cg] = v;
    if (outT) outT[base + (size_t)cg * 256 + rg] = v;
  }
}

// ---------------------------------------------------------------------------
// QK^T + row softmax over 256 landmarks (used for attn1 and attn2).
// Block: 16 rows x 256 cols; 8 waves each produce a 16x32 logit strip.
// ---------------------------------------------------------------------------
__global__ __launch_bounds__(256) void attn_qk_softmax_kernel(
    const _Float16* __restrict__ Q, long long sQ,
    const _Float16* __restrict__ KL, long long sK,
    _Float16* __restrict__ O16, long long sO,
    float* __restrict__ O32, long long sO32) {
  __shared__ float Ls[16][256];
  __shared__ float red[16][16];
  __shared__ float rowstat[16];
  int tid = threadIdx.x, w = tid >> 5, lane = tid & 31;
  int c = lane & 15, hi = lane >> 4;
  int bh = blockIdx.z;
  const _Float16* Qb = Q + (size_t)bh * sQ + (size_t)blockIdx.x * 16 * 64;
  const _Float16* Kb = KL + (size_t)bh * sK;
  v8f acc[2] = {};
  for (int k0 = 0; k0 < 64; k0 += 32) {
    v16h af = load_a_frag(Qb, 64, k0, lane);
#pragma unroll
    for (int t = 0; t < 2; ++t)
      acc[t] = wmma16(
          af, load_bt_frag(Kb + (size_t)(w * 32 + t * 16) * 64, 64, k0, lane),
          acc[t]);
  }
#pragma unroll
  for (int t = 0; t < 2; ++t)
#pragma unroll
    for (int i = 0; i < 8; ++i)
      Ls[i + 8 * hi][w * 32 + t * 16 + c] = acc[t][i];
  __syncthreads();

  int r = tid >> 4, c0 = tid & 15;
  float mx = -1e30f;
#pragma unroll
  for (int j = 0; j < 16; ++j) mx = fmaxf(mx, Ls[r][c0 + 16 * j]);
  red[r][c0] = mx; __syncthreads();
  if (tid < 16) {
    float m = -1e30f;
#pragma unroll
    for (int j = 0; j < 16; ++j) m = fmaxf(m, red[tid][j]);
    rowstat[tid] = m;
  }
  __syncthreads();
  float rm = rowstat[r];
  float s = 0.f;
#pragma unroll
  for (int j = 0; j < 16; ++j) {
    float e = expf(Ls[r][c0 + 16 * j] - rm);
    Ls[r][c0 + 16 * j] = e;
    s += e;
  }
  __syncthreads();
  red[r][c0] = s; __syncthreads();
  if (tid < 16) {
    float sm = 0.f;
#pragma unroll
    for (int j = 0; j < 16; ++j) sm += red[tid][j];
    rowstat[tid] = sm;
  }
  __syncthreads();
  float inv = 1.0f / rowstat[r];
  size_t ro16 = (size_t)bh * sO + (size_t)(blockIdx.x * 16 + r) * 256;
#pragma unroll
  for (int j = 0; j < 16; ++j) {
    float p = Ls[r][c0 + 16 * j] * inv;
    O16[ro16 + c0 + 16 * j] = (_Float16)p;
    if (O32)
      O32[(size_t)bh * sO32 + (size_t)(blockIdx.x * 16 + r) * 256 + c0 + 16 * j] = p;
  }
}

// ---------------------------------------------------------------------------
// Fused attn3@v: streaming two-pass softmax over N=4096 (chunks of 128),
// P staged as f16 in LDS and fed directly to WMMA; output stored transposed.
// ---------------------------------------------------------------------------
__global__ __launch_bounds__(256) void attn3v_kernel(
    const _Float16* __restrict__ QL, const _Float16* __restrict__ K,
    const _Float16* __restrict__ VT, _Float16* __restrict__ A3VT) {
  __shared__ __align__(32) float Ls[16][128];
  __shared__ __align__(32) _Float16 Pm[16][128];
  __shared__ float red[16][16], red2[16][16];
  __shared__ float rmax[16], rsum[16];
  int tid = threadIdx.x, w = tid >> 5, lane = tid & 31;
  int c = lane & 15, hi = lane >> 4;
  int bh = blockIdx.z, rb = blockIdx.x;  // 16 landmark-row blocks
  const _Float16* Qb = QL + ((size_t)bh * 256 + rb * 16) * 64;
  const _Float16* Kb = K + (size_t)bh * 4096 * 64;
  const _Float16* Vb = VT + (size_t)bh * 64 * 4096;
  int r16 = tid >> 4, c0 = tid & 15;

  // pass 1: streaming row max / sum-exp
  float m = -1e30f, s = 0.f;
  for (int ch = 0; ch < 4096; ch += 128) {
    v8f acc = {};
    for (int k0 = 0; k0 < 64; k0 += 32)
      acc = wmma16(load_a_frag(Qb, 64, k0, lane),
                   load_bt_frag(Kb + (size_t)(ch + w * 16) * 64, 64, k0, lane), acc);
#pragma unroll
    for (int i = 0; i < 8; ++i) Ls[i + 8 * hi][w * 16 + c] = acc[i];
    __syncthreads();
    float cm = -1e30f;
#pragma unroll
    for (int jj = 0; jj < 8; ++jj) cm = fmaxf(cm, Ls[r16][c0 * 8 + jj]);
    float mn = fmaxf(m, cm);
    float cs = 0.f;
#pragma unroll
    for (int jj = 0; jj < 8; ++jj) cs += expf(Ls[r16][c0 * 8 + jj] - mn);
    s = s * expf(m - mn) + cs;
    m = mn;
    __syncthreads();
  }
  red[r16][c0] = m; red2[r16][c0] = s; __syncthreads();
  if (tid < 16) {
    float M = -1e30f;
#pragma unroll
    for (int j = 0; j < 16; ++j) M = fmaxf(M, red[tid][j]);
    float S = 0.f;
#pragma unroll
    for (int j = 0; j < 16; ++j) S += red2[tid][j] * expf(red[tid][j] - M);
    rmax[tid] = M; rsum[tid] = S;
  }
  __syncthreads();
  float RM = rmax[r16], RS = rsum[r16];

  // pass 2: recompute logits, normalize, accumulate P @ V (waves 0..3 own d-tiles)
  v8f accO = {};
  for (int ch = 0; ch < 4096; ch += 128) {
    v8f acc = {};
    for (int k0 = 0; k0 < 64; k0 += 32)
      acc = wmma16(load_a_frag(Qb, 64, k0, lane),
                   load_bt_frag(Kb + (size_t)(ch + w * 16) * 64, 64, k0, lane), acc);
#pragma unroll
    for (int i = 0; i < 8; ++i) Ls[i + 8 * hi][w * 16 + c] = acc[i];
    __syncthreads();
#pragma unroll
    for (int jj = 0; jj < 8; ++jj)
      Pm[r16][c0 * 8 + jj] = (_Float16)(expf(Ls[r16][c0 * 8 + jj] - RM) / RS);
    __syncthreads();
    if (w < 4) {
      for (int k0 = 0; k0 < 128; k0 += 32)
        accO = wmma16(load_a_frag(&Pm[0][0], 128, k0, lane),
                      load_bt_frag(Vb + (size_t)(w * 16) * 4096 + ch, 4096, k0, lane),
                      accO);
    }
    __syncthreads();
  }
  if (w < 4) {
#pragma unroll
    for (int i = 0; i < 8; ++i)
      A3VT[(size_t)bh * 64 * 256 + (size_t)(w * 16 + c) * 256 + rb * 16 + i + 8 * hi] =
          (_Float16)accO[i];
  }
}

// ---------------------------------------------------------------------------
// Host launch
// ---------------------------------------------------------------------------
extern "C" void kernel_launch(void* const* d_in, const int* in_sizes, int n_in,
                              void* d_out, int out_size, void* d_ws, size_t ws_size,
                              hipStream_t stream) {
  const float* x    = (const float*)d_in[0];  // [4,4096,512]
  const float* Wqkv = (const float*)d_in[1];  // [512,1536]
  const float* Wout = (const float*)d_in[2];  // [512,512]
  const float* bout = (const float*)d_in[3];  // [512]
  const float* Wres = (const float*)d_in[4];  // [8,1,33,1]

  const size_t NE_XH  = 16384ull * 512;
  const size_t NE_QH  = 32ull * 4096 * 64;
  const size_t NE_QLH = 32ull * 256 * 64;
  const size_t NE_A2  = 32ull * 256 * 256;
  const size_t NE_A1  = 32ull * 4096 * 256;

  char* wsb = (char*)d_ws;
  size_t off = 0;
  auto arena = [&](size_t bytes) -> void* {
    void* p = wsb + off;
    off = (off + bytes + 255) & ~(size_t)255;
    return p;
  };
  _Float16* xh     = (_Float16*)arena(NE_XH * 2);
  _Float16* wqkvT  = (_Float16*)arena(1536ull * 512 * 2);
  _Float16* woutT  = (_Float16*)arena(512ull * 512 * 2);
  _Float16* qh     = (_Float16*)arena(NE_QH * 2);
  _Float16* kh     = (_Float16*)arena(NE_QH * 2);
  _Float16* vh     = (_Float16*)arena(NE_QH * 2);
  _Float16* vTh    = (_Float16*)arena(NE_QH * 2);
  _Float16* qlh    = (_Float16*)arena(NE_QLH * 2);
  _Float16* klh    = (_Float16*)arena(NE_QLH * 2);
  _Float16* a2h    = (_Float16*)arena(NE_A2 * 2);
  float*    a2f    = (float*)arena(NE_A2 * 4);
  _Float16* a1h    = (_Float16*)arena(NE_A1 * 2);
  float*    zaf    = (float*)arena(NE_A2 * 4);
  float*    zaTf   = (float*)arena(NE_A2 * 4);
  float*    zbf    = (float*)arena(NE_A2 * 4);
  float*    zbTf   = (float*)arena(NE_A2 * 4);
  float*    xzf    = (float*)arena(NE_A2 * 4);
  float*    xzTf   = (float*)arena(NE_A2 * 4);
  float*    t1Tf   = (float*)arena(NE_A2 * 4);
  float*    t2Tf   = (float*)arena(NE_A2 * 4);
  float*    t3Tf   = (float*)arena(NE_A2 * 4);
  _Float16* pinvTh = (_Float16*)arena(NE_A2 * 2);
  _Float16* a3vT   = (_Float16*)arena(32ull * 64 * 256 * 2);
  _Float16* Th     = (_Float16*)arena(NE_A1 * 2);
  float*    Of     = (float*)arena(NE_QH * 4);
  float*    scal   = (float*)arena(256);
  _Float16* hidden = xh;  // x_f16 dead after QKV GEMM; reuse for hidden [16384,512]

  // 1) conversions
  f32_to_f16_kernel<<<(unsigned)((NE_XH + 255) / 256), 256, 0, stream>>>(x, xh, (int)NE_XH);
  transpose_f32_to_f16_kernel<<<(512 * 1536 + 255) / 256, 256, 0, stream>>>(Wqkv, wqkvT, 512, 1536);
  transpose_f32_to_f16_kernel<<<(512 * 512 + 255) / 256, 256, 0, stream>>>(Wout, woutT, 512, 512);
  init_scal_kernel<<<1, 32, 0, stream>>>(scal);

  // 2) QKV projection (q scaled)
  gemm_qkv_kernel<<<dim3(24, 128, 1), 256, 0, stream>>>(xh, wqkvT, qh, kh, vh, vTh);

  // 3) landmarks
  landmark_kernel<<<(unsigned)(NE_QLH / 256), 256, 0, stream>>>(qh, qlh);
  landmark_kernel<<<(unsigned)(NE_QLH / 256), 256, 0, stream>>>(kh, klh);

  // 4) attn2 = softmax(qland @ kland^T)  (f16 + f32 copies)
  attn_qk_softmax_kernel<<<dim3(16, 1, 32), 256, 0, stream>>>(
      qlh, 256LL * 64, klh, 256LL * 64, a2h, 256LL * 256, a2f, 256LL * 256);

  // 5) Moore-Penrose init: z0 = attn2^T / (max rowsum * max colsum)
  attn2_reduce_kernel<<<32, 256, 0, stream>>>(a2f, scal);
  pinv_init_f32_kernel<<<dim3(1, 1, 32), 256, 0, stream>>>(a2f, scal, zaf, zaTf);

  // 6) Newton-Schulz iterations in f32 WMMA (all NT matmuls; B kept transposed)
  float *cur = zaf, *curT = zaTf, *nxt = zbf, *nxtT = zbTf;
  for (int it = 0; it < 6; ++it) {
    mm256_f32_kernel<<<dim3(4, 8, 32), 256, 0, stream>>>(a2f, curT, xzf, xzTf, 1.f, 0.f);
    diag_axpy_f32_kernel<<<dim3(1, 1, 32), 256, 0, stream>>>(xzTf, t1Tf, -1.f, 7.f);
    mm256_f32_kernel<<<dim3(4, 8, 32), 256, 0, stream>>>(xzf, t1Tf, (float*)nullptr, t2Tf, -1.f, 15.f);
    mm256_f32_kernel<<<dim3(4, 8, 32), 256, 0, stream>>>(xzf, t2Tf, (float*)nullptr, t3Tf, -1.f, 13.f);
    mm256_f32_kernel<<<dim3(4, 8, 32), 256, 0, stream>>>(cur, t3Tf, nxt, nxtT, 0.25f, 0.f);
    float* t;
    t = cur; cur = nxt; nxt = t;
    t = curT; curT = nxtT; nxtT = t;
  }
  // pinv^T -> f16 for the big attn1 @ pinv GEMM
  f32_to_f16_kernel<<<(unsigned)(NE_A2 / 256), 256, 0, stream>>>(curT, pinvTh, (int)NE_A2);

  // 7) attn1 = softmax(q @ kland^T)
  attn_qk_softmax_kernel<<<dim3(256, 1, 32), 256, 0, stream>>>(
      qh, 4096LL * 64, klh, 256LL * 64, a1h, 4096LL * 256, (float*)nullptr, 0LL);

  // 8) attn3v^T = (softmax(qland @ k^T) @ v)^T  (fused streaming softmax)
  attn3v_kernel<<<dim3(16, 1, 32), 256, 0, stream>>>(qlh, kh, vTh, a3vT);

  // 9) T = attn1 @ pinv   (B = pinv^T)
  gemm_nt_kernel<_Float16, false><<<dim3(4, 32, 32), 256, 0, stream>>>(
      a1h, 256, 4096LL * 256, pinvTh, 256, 65536LL, Th, 256, 4096LL * 256, 256,
      (const float*)nullptr);

  // 10) O = T @ attn3v    (B = attn3v^T)
  gemm_nt_kernel<float, false><<<dim3(1, 32, 32), 256, 0, stream>>>(
      Th, 256, 4096LL * 256, a3vT, 256, 64LL * 256, Of, 64, 4096LL * 64, 256,
      (const float*)nullptr);

  // 11) depthwise conv residual + reshape to hidden f16
  residual_kernel<<<(unsigned)(NE_QH / 256), 256, 0, stream>>>(Of, vh, Wres, hidden);

  // 12) out = hidden @ W_out + b_out  (f32 output)
  gemm_nt_kernel<float, true><<<dim3(8, 128, 1), 256, 0, stream>>>(
      hidden, 512, 0LL, woutT, 512, 0LL, (float*)d_out, 512, 0LL, 512, bout);
}
